// Decoder_SR_55147380081265
// MI455X (gfx1250) — matrix operand order repaired and checked by reading, not compiled, not run on hardware
//
#include <hip/hip_runtime.h>

typedef __attribute__((ext_vector_type(16))) __bf16 v16bf;
typedef __attribute__((ext_vector_type(8)))  float  v8f;
typedef __attribute__((ext_vector_type(8)))  unsigned v8u;

#define N_NODES 100000
#define N_EDGES 3200000
#define FDIM    128

// pack two f32 into one dword of bf16 (round-to-nearest-even)
__device__ __forceinline__ unsigned bfpack(float x, float y) {
    unsigned ux = __builtin_bit_cast(unsigned, x);
    unsigned uy = __builtin_bit_cast(unsigned, y);
    ux = (ux + 0x7FFFu + ((ux >> 16) & 1u)) >> 16;
    uy = (uy + 0x7FFFu + ((uy >> 16) & 1u)) >> 16;
    return (ux & 0xFFFFu) | ((uy & 0xFFFFu) << 16);
}

// Native CDNA5 f32 atomic add (no-return form, STOREcnt-tracked)
__device__ __forceinline__ void atomic_add_f32(float* p, float v) {
    asm volatile("global_atomic_add_f32 %0, %1, off" :: "v"(p), "v"(v) : "memory");
}

// One-shot: W[K x 128] f32 -> bf16 B-fragments in a contiguous-k 32x16 WMMA VGPR
// layout. Wbf dword index = ((q*8 + tile)*32 + lane)*8 + i holds
// B[q*32 + 2i + 16*(lane>>4)][tile*16 + (lane&15)] and the k+1 row packed hi.
__global__ void __launch_bounds__(256)
convert_W(const float* __restrict__ W, unsigned* __restrict__ Wbf, int K) {
    const int idx  = blockIdx.x * 256 + threadIdx.x;   // grid exact: K*64/256
    const int i    = idx & 7;
    const int lane = (idx >> 3) & 31;
    const int tile = (idx >> 8) & 7;
    const int q    = idx >> 11;
    const int col  = tile * 16 + (lane & 15);
    const int kb   = q * 32 + 2 * i + 16 * (lane >> 4);
    Wbf[idx] = bfpack(W[kb * FDIM + col], W[(kb + 1) * FDIM + col]);
}

// C[M x 128] = A[M x K] @ W[K x 128]. Block = 8 waves; wave w owns column
// tile w; block owns 16 rows (shared A staged once in LDS by all 8 waves).
// Grid is exact (M%16==0) so EXEC is all-ones as WMMA requires.
template <int K>
__global__ void __launch_bounds__(256)
gemm_wmma(const float* __restrict__ A, const unsigned* __restrict__ Wbf,
          float* __restrict__ C) {
    constexpr int NCH  = K / 32;   // k-chunks
    constexpr int LSTR = 12;       // padded dwords per lane-fragment slot
    __shared__ unsigned ldsA[NCH * 32 * LSTR];

    const int  tid     = threadIdx.x;
    const long rowbase = (long)blockIdx.x * 16 * K;

    // Cooperative stage: 16 x K f32 tile -> bf16 A-fragments in LDS.
#pragma unroll
    for (int it = 0; it < (16 * (K / 2)) / 256; ++it) {
        const int d   = it * 256 + tid;
        const int row = d / (K / 2);
        const int kp  = d % (K / 2);
        const float2 af = *(const float2*)(A + rowbase + (long)row * K + 2 * kp);
        const unsigned dw = bfpack(af.x, af.y);
        const int k    = 2 * kp;
        const int q    = k >> 5;
        const int kk   = k & 31;
        const int half = (kk >> 4) & 1;            // lane group for this kk
        const int kkp  = kk - 16 * half;           // {0..15}
        const int i    = kkp >> 1;
        ldsA[(q * 32 + row + 16 * half) * LSTR + i] = dw;
    }
    __syncthreads();

    const int lane = tid & 31;
    const int wave = tid >> 5;
    v8f acc = {};
#pragma unroll
    for (int q = 0; q < NCH; ++q) {
        const int aoff = (q * 32 + lane) * LSTR;
        const uint4 a0 = *(const uint4*)&ldsA[aoff];
        const uint4 a1 = *(const uint4*)&ldsA[aoff + 4];
        const int boff = ((q * 8 + wave) * 32 + lane) * 8;
        const uint4 b0 = *(const uint4*)&Wbf[boff];
        const uint4 b1 = *(const uint4*)&Wbf[boff + 4];
        v8u au, bu;
        au[0] = a0.x; au[1] = a0.y; au[2] = a0.z; au[3] = a0.w;
        au[4] = a1.x; au[5] = a1.y; au[6] = a1.z; au[7] = a1.w;
        bu[0] = b0.x; bu[1] = b0.y; bu[2] = b0.z; bu[3] = b0.w;
        bu[4] = b1.x; bu[5] = b1.y; bu[6] = b1.z; bu[7] = b1.w;
        acc = __builtin_amdgcn_wmma_f32_16x16x32_bf16(
            false, __builtin_bit_cast(v16bf, au),
            false, __builtin_bit_cast(v16bf, bu),
            (short)0, acc, false, false);
    }

    // C/D 16x16 f32 layout: lane -> N = lane&15, VGPR j -> M = j + 8*(lane>>4)
    const int half = lane >> 4, l16 = lane & 15;
    float* crow = C + (long)blockIdx.x * 16 * FDIM + wave * 16 + l16;
#pragma unroll
    for (int j = 0; j < 8; ++j) {
        crow[(long)(j + 8 * half) * FDIM] = acc[j];
    }
}

// out[n][f] = bias[f]  (also deterministically clears poison)
__global__ void __launch_bounds__(256)
init_bias(float* __restrict__ o, const float* __restrict__ b) {
    const long i = (long)blockIdx.x * blockDim.x + threadIdx.x;
    o[i] = b[i & (FDIM - 1)];
}

// One wave per edge: out[r][:] += v * S[c][:], 4 features per lane.
__global__ void __launch_bounds__(256)
spmm_scatter(const int* __restrict__ rows, const int* __restrict__ cols,
             const float* __restrict__ vals, const float* __restrict__ S,
             float* __restrict__ out, int E) {
    const int wid  = (int)(((unsigned)blockIdx.x * blockDim.x + threadIdx.x) >> 5);
    const int lane = threadIdx.x & 31;
    if (wid >= E) return;
    const int   r = __builtin_amdgcn_readfirstlane(rows[wid]);
    const int   c = __builtin_amdgcn_readfirstlane(cols[wid]);
    const float v = vals[wid];
    const float4 s = *(const float4*)(S + (long)c * FDIM + lane * 4);
    float* dst = out + (long)r * FDIM + lane * 4;
    atomic_add_f32(dst + 0, v * s.x);
    atomic_add_f32(dst + 1, v * s.y);
    atomic_add_f32(dst + 2, v * s.z);
    atomic_add_f32(dst + 3, v * s.w);
}

__global__ void __launch_bounds__(256)
relu_k(float* __restrict__ o) {
    const long i = (long)blockIdx.x * blockDim.x + threadIdx.x;
    const float v = o[i];
    o[i] = v > 0.0f ? v : 0.0f;
}

extern "C" void kernel_launch(void* const* d_in, const int* in_sizes, int n_in,
                              void* d_out, int out_size, void* d_ws, size_t ws_size,
                              hipStream_t stream) {
    const float* x    = (const float*)d_in[0];
    const int*   rows = (const int*)d_in[1];            // edge_index[0]
    const int*   cols = rows + N_EDGES;                 // edge_index[1]
    const float* vals = (const float*)d_in[2];
    const float* W1   = (const float*)d_in[3];
    const float* b1   = (const float*)d_in[4];
    const float* W2   = (const float*)d_in[5];
    const float* b2   = (const float*)d_in[6];
    const float* W3   = (const float*)d_in[7];
    const float* b3   = (const float*)d_in[8];

    float* out = (float*)d_out;                         // [4, N, 128]
    float* S   = (float*)d_ws;                          // support scratch [N,128]
    const long slice = (long)N_NODES * FDIM;
    // Pre-converted weight fragments live after S in scratch.
    unsigned* Wbf1 = (unsigned*)(S + slice);            // 256*64 dwords
    unsigned* Wbf2 = Wbf1 + 256 * 64;                   // 128*64 dwords
    unsigned* Wbf3 = Wbf2 + 128 * 64;                   // 128*64 dwords
    // Layer-1 hidden state staged in the last output slice; fully consumed
    // (layer-2 GEMM) before that slice is overwritten at t=3.
    float* H = out + 3 * slice;

    const int nfBlocks = (int)(slice / 256);            // 50000, exact
    const int eBlocks  = N_EDGES / 8;                   // 8 waves/block, exact
    const dim3 blk(256);

    convert_W<<<256 * 64 / 256, blk, 0, stream>>>(W1, Wbf1, 256);
    convert_W<<<128 * 64 / 256, blk, 0, stream>>>(W2, Wbf2, 128);
    convert_W<<<128 * 64 / 256, blk, 0, stream>>>(W3, Wbf3, 128);

    auto post = [&](const float* bl, float* hout) {
        init_bias<<<nfBlocks, blk, 0, stream>>>(hout, bl);
        spmm_scatter<<<eBlocks, blk, 0, stream>>>(rows, cols, vals, S, hout, N_EDGES);
        relu_k<<<nfBlocks, blk, 0, stream>>>(hout);
    };

    // layer 1 (K=256)
    gemm_wmma<256><<<N_NODES / 16, blk, 0, stream>>>(x, Wbf1, S);
    post(b1, H);
    // layer 2 (K=128) -> preds[0]
    gemm_wmma<128><<<N_NODES / 16, blk, 0, stream>>>(H, Wbf2, S);
    post(b2, out + 0 * slice);
    // t = 1..3 share W3
    gemm_wmma<128><<<N_NODES / 16, blk, 0, stream>>>(out + 0 * slice, Wbf3, S);
    post(b3, out + 1 * slice);
    gemm_wmma<128><<<N_NODES / 16, blk, 0, stream>>>(out + 1 * slice, Wbf3, S);
    post(b3, out + 2 * slice);
    gemm_wmma<128><<<N_NODES / 16, blk, 0, stream>>>(out + 2 * slice, Wbf3, S);
    post(b3, out + 3 * slice);                          // overwrites H last
}
